// PCN_encoder_25744033972931
// MI455X (gfx1250) — matrix operand order repaired and checked by reading, not compiled, not run on hardware
//
#include <hip/hip_runtime.h>
#include <hip/hip_bf16.h>

// ---------------------------------------------------------------------------
// PCN encoder for MI455X (gfx1250, wave32).
// All GEMMs run on v_wmma_f32_16x16x32_bf16 with fp32 accumulation.
// Every operand fragment is a per-lane contiguous 16B load (no scalar packing):
// bf16 activations are kept point-major [B][N][C], attention scores are
// transposed+converted to bf16 once per SA block.
// Workspace requirement: ~264 MB (att fp32 134 MB + attT bf16 67 MB dominate;
// per-batch slices are L2-resident on the 192 MB L2).
// ---------------------------------------------------------------------------

typedef __attribute__((ext_vector_type(16))) __bf16 v16bf;
typedef __attribute__((ext_vector_type(8)))  float  v8f;
typedef unsigned short u16;
typedef unsigned int   u32;

union Frag {
    v16bf v;
    u16   u[16];
    uint4 q[2];
};

__device__ __forceinline__ u16 f2b(float x) {
    u32 u = __builtin_bit_cast(u32, x);
    u32 r = u + 0x7FFFu + ((u >> 16) & 1u);   // round-to-nearest-even
    return (u16)(r >> 16);
}

// Epilogue modes
enum { EPI_ENERGY = 0, EPI_QT = 1, EPI_BIAS_BF16 = 2, EPI_ATT_T = 3,
       EPI_BNRELU = 4, EPI_BNRELU_RES = 5 };

// ---------------------------------------------------------------------------
// bf16 WMMA GEMM: D[M,N] = A[M,K] * Bt[N,K]^T, fp32 accumulate.
//   A  : bf16 row-major  (A[m][k], leading dim lda)
//   Bt : bf16 "column-major B" = row-major [N][K] (Bt[n][k], leading dim ldb)
// Block = 128 threads = 4 waves; each wave owns a 32x64 output tile
// (2 A-frags x 4 B-frags -> 8 v_wmma per 32-wide K step).
// Grid: (N/128, M/64, BATCH).  M % 64 == 0, N % 128 == 0, K % 32 == 0.
// No LDS staging: operands are L2-resident and all fragment loads are
// per-lane contiguous uint4 loads.
// ---------------------------------------------------------------------------
template <int MODE>
__global__ __launch_bounds__(128)
void gemm_wmma(const u16* __restrict__ A, long long sA, int lda,
               const u16* __restrict__ Bt, long long sB, int ldb,
               float* __restrict__ outF, u16* __restrict__ outH,
               long long sO, int ldo,
               const float* __restrict__ p0,     // bias (XV) or BN scale
               const float* __restrict__ p1,     // BN shift
               const float* __restrict__ resid, long long sR,
               const float* __restrict__ colsum, // 1e-9 + sum_n att[n][m]
               int K)
{
    const int lane = threadIdx.x & 31;
    const int wid  = threadIdx.x >> 5;
    const int wm = wid >> 1, wn = wid & 1;
    const int b  = blockIdx.z;
    const int row0 = blockIdx.y * 64 + wm * 32;
    const int col0 = blockIdx.x * 128 + wn * 64;

    const u16* Ab  = A  + (size_t)b * sA;
    const u16* Bb  = Bt + (size_t)b * sB;

    // A fragment lane mapping (16-bit A 16x32, ISA 7.12.2):
    //   lanes 0-15:  M=lane,    K = 0..7  then 16..23
    //   lanes 16-31: M=lane-16, K = 8..15 then 24..31
    const int ar  = lane & 15;
    const int ako = (lane >> 4) * 8;
    // B fragment lane mapping (16-bit B 32x16):
    //   lanes 0-15: col=lane, K=0..7 then 16..23 ; lanes 16-31: col=lane-16, K=8..15 then 24..31
    const int bc  = lane & 15;
    const int bkb = (lane >> 4) * 8;

    v8f acc[2][4];
#pragma unroll
    for (int i = 0; i < 2; ++i)
#pragma unroll
        for (int j = 0; j < 4; ++j)
#pragma unroll
            for (int e = 0; e < 8; ++e) acc[i][j][e] = 0.0f;

    for (int k = 0; k < K; k += 32) {
        Frag a[2], bf[4];
#pragma unroll
        for (int i = 0; i < 2; ++i) {
            const u16* p = Ab + (size_t)(row0 + i * 16 + ar) * lda + k + ako;
            a[i].q[0] = *(const uint4*)(p);
            a[i].q[1] = *(const uint4*)(p + 16);
        }
#pragma unroll
        for (int j = 0; j < 4; ++j) {
            const u16* p = Bb + (size_t)(col0 + j * 16 + bc) * ldb + k + bkb;
            bf[j].q[0] = *(const uint4*)(p);
            bf[j].q[1] = *(const uint4*)(p + 16);
        }
#pragma unroll
        for (int i = 0; i < 2; ++i)
#pragma unroll
            for (int j = 0; j < 4; ++j)
                acc[i][j] = __builtin_amdgcn_wmma_f32_16x16x32_bf16(
                    false, a[i].v, false, bf[j].v, (short)0, acc[i][j],
                    false, false);
    }

    // ---- Epilogue.  C/D layout: VGPR v, lanes 0-15: M=v, N=lane;
    //                             lanes 16-31: M=v+8, N=lane-16.
    const int cIn   = lane & 15;
    const int rb    = (lane >> 4) * 8;
    const int Ncols = gridDim.x * 128;   // logical N (colsum indexing)
    const int Mtot  = gridDim.y * 64;    // logical M (transposed-store ld)

#pragma unroll
    for (int i = 0; i < 2; ++i) {
#pragma unroll
        for (int j = 0; j < 4; ++j) {
            const int R0 = row0 + i * 16 + rb;
            const int C0 = col0 + j * 16 + cIn;
            v8f a = acc[i][j];
            if (MODE == EPI_ENERGY) {
                float* o = outF + (size_t)b * sO + C0;
#pragma unroll
                for (int v = 0; v < 8; ++v) o[(size_t)(R0 + v) * ldo] = a[v];
            } else if (MODE == EPI_QT) {
                // q transposed: qT[n][c] bf16, one 16B store per lane
                u16 tmp[8];
#pragma unroll
                for (int v = 0; v < 8; ++v) tmp[v] = f2b(a[v]);
                *(uint4*)(outH + (size_t)b * sO + (size_t)C0 * Mtot + R0) =
                    *(const uint4*)tmp;
            } else if (MODE == EPI_BIAS_BF16) {
                // xv row-major bf16 [C][N] (A operand of the att matmul)
                u16* o = outH + (size_t)b * sO + C0;
#pragma unroll
                for (int v = 0; v < 8; ++v)
                    o[(size_t)(R0 + v) * ldo] = f2b(a[v] + p0[R0 + v]);
            } else if (MODE == EPI_ATT_T) {
                // t = h - (xv@att)/(1e-9+colsum[m]); store transposed bf16
                const float inv = 1.0f / colsum[(size_t)b * Ncols + C0];
                const float* r = resid + (size_t)b * sR + C0;
                u16 tmp[8];
#pragma unroll
                for (int v = 0; v < 8; ++v)
                    tmp[v] = f2b(r[(size_t)(R0 + v) * ldo] - a[v] * inv);
                *(uint4*)(outH + (size_t)b * sO + (size_t)C0 * Mtot + R0) =
                    *(const uint4*)tmp;
            } else { // EPI_BNRELU / EPI_BNRELU_RES
                float* oF = outF + (size_t)b * sO + C0;
                float y[8];
#pragma unroll
                for (int v = 0; v < 8; ++v) {
                    y[v] = fmaxf(0.0f, a[v] * p0[R0 + v] + p1[R0 + v]);
                    if (MODE == EPI_BNRELU_RES)
                        y[v] += resid[(size_t)b * sR + (size_t)(R0 + v) * ldo + C0];
                    oF[(size_t)(R0 + v) * ldo] = y[v];
                }
                if (outH) {  // transposed bf16 mirror [N][M]
                    u16 tmp[8];
#pragma unroll
                    for (int v = 0; v < 8; ++v) tmp[v] = f2b(y[v]);
                    *(uint4*)(outH + (size_t)b * sO + (size_t)C0 * Mtot + R0) =
                        *(const uint4*)tmp;
                }
            }
        }
    }
}

// ---------------------------------------------------------------------------
// Small support kernels
// ---------------------------------------------------------------------------
__global__ void cvt_k(const float* __restrict__ s, u16* __restrict__ d, int n) {
    int i = blockIdx.x * 256 + threadIdx.x;
    if (i < n) d[i] = f2b(s[i]);
}

__global__ void bnprep_k(const float* __restrict__ bias, const float* __restrict__ g,
                         const float* __restrict__ beta, const float* __restrict__ m,
                         const float* __restrict__ v, float* __restrict__ s,
                         float* __restrict__ sh, int C) {
    int c = blockIdx.x * 256 + threadIdx.x;
    if (c < C) {
        float sc = g[c] * rsqrtf(v[c] + 1e-5f);
        s[c]  = sc;
        sh[c] = (bias[c] - m[c]) * sc + beta[c];
    }
}

// Fused conv1(78->32)+BN+ReLU and conv2(32->64)+BN+ReLU per point.
// Writes fp32 [B][64][N] and bf16 point-major [B][N][64].
__global__ __launch_bounds__(256)
void conv12_k(const float* __restrict__ x,
              const float* __restrict__ W1, const float* __restrict__ b1,
              const float* __restrict__ g1, const float* __restrict__ be1,
              const float* __restrict__ m1, const float* __restrict__ v1,
              const float* __restrict__ W2, const float* __restrict__ b2,
              const float* __restrict__ g2, const float* __restrict__ be2,
              const float* __restrict__ m2, const float* __restrict__ v2,
              float* __restrict__ outF, u16* __restrict__ outT, int Nn)
{
    __shared__ float sW1[32 * 78], sW2[64 * 32];
    __shared__ float s1[32], sh1[32], s2[64], sh2[64];
    const int t = threadIdx.x;
    for (int i = t; i < 32 * 78; i += 256) sW1[i] = W1[i];
    for (int i = t; i < 64 * 32; i += 256) sW2[i] = W2[i];
    if (t < 32) { float sc = g1[t] * rsqrtf(v1[t] + 1e-5f); s1[t] = sc; sh1[t] = (b1[t] - m1[t]) * sc + be1[t]; }
    if (t < 64) { float sc = g2[t] * rsqrtf(v2[t] + 1e-5f); s2[t] = sc; sh2[t] = (b2[t] - m2[t]) * sc + be2[t]; }
    __syncthreads();

    const int n = blockIdx.x * 256 + t, b = blockIdx.y;
    const float* xp = x + (size_t)b * 78 * Nn + n;
    float xin[78];
#pragma unroll
    for (int c = 0; c < 78; ++c) xin[c] = xp[(size_t)c * Nn];
    float h1[32];
#pragma unroll
    for (int o = 0; o < 32; ++o) {
        float a = 0.0f;
#pragma unroll
        for (int c = 0; c < 78; ++c) a += sW1[o * 78 + c] * xin[c];
        h1[o] = fmaxf(0.0f, a * s1[o] + sh1[o]);
    }
    float* of = outF + (size_t)b * 64 * Nn + n;
    u16*   ot = outT + ((size_t)b * Nn + n) * 64;
#pragma unroll
    for (int o = 0; o < 64; ++o) {
        float a = 0.0f;
#pragma unroll
        for (int c = 0; c < 32; ++c) a += sW2[o * 32 + c] * h1[c];
        float y = fmaxf(0.0f, a * s2[o] + sh2[o]);
        of[(size_t)o * Nn] = y;
        ot[o] = f2b(y);
    }
}

// Row softmax over last dim, fp32 in place.  Grid (N, B), block 256.
__global__ __launch_bounds__(256)
void softmax_rows(float* __restrict__ att, int Nn)
{
    const long long row = (long long)blockIdx.y * gridDim.x + blockIdx.x;
    float* p = att + row * Nn;
    __shared__ float red[256];
    const int t = threadIdx.x;
    float vals[8];
    float mx = -3.4e38f;
#pragma unroll
    for (int i = 0; i < 8; ++i) { vals[i] = p[t + i * 256]; mx = fmaxf(mx, vals[i]); }
    red[t] = mx; __syncthreads();
    for (int s = 128; s > 0; s >>= 1) { if (t < s) red[t] = fmaxf(red[t], red[t + s]); __syncthreads(); }
    mx = red[0]; __syncthreads();
    float sm = 0.0f;
#pragma unroll
    for (int i = 0; i < 8; ++i) { vals[i] = __expf(vals[i] - mx); sm += vals[i]; }
    red[t] = sm; __syncthreads();
    for (int s = 128; s > 0; s >>= 1) { if (t < s) red[t] += red[t + s]; __syncthreads(); }
    const float inv = 1.0f / red[0];
#pragma unroll
    for (int i = 0; i < 8; ++i) p[t + i * 256] = vals[i] * inv;
}

// Column sums over n (renorm denominator).  Grid (N/256, B).
__global__ void colsum_k(const float* __restrict__ att, float* __restrict__ cs, int Nn)
{
    const int m = blockIdx.x * 256 + threadIdx.x, b = blockIdx.y;
    const float* p = att + (size_t)b * Nn * Nn + m;
    float s = 0.0f;
    for (int n = 0; n < Nn; ++n) s += p[(size_t)n * Nn];
    cs[(size_t)b * Nn + m] = s + 1e-9f;
}

// attT[m][n] = bf16(att[n][m]).  LDS-tiled transpose+convert.
__global__ __launch_bounds__(256)
void attT_k(const float* __restrict__ att, u16* __restrict__ attT, int Nn)
{
    __shared__ float tile[32][33];
    const int b = blockIdx.z;
    const int n0 = blockIdx.x * 32, m0 = blockIdx.y * 32;
    const int tx = threadIdx.x, ty = threadIdx.y;   // 32 x 8
    const float* p = att + (size_t)b * Nn * Nn;
#pragma unroll
    for (int s = 0; s < 4; ++s)
        tile[ty + 8 * s][tx] = p[(size_t)(n0 + ty + 8 * s) * Nn + m0 + tx];
    __syncthreads();
    u16* o = attT + (size_t)b * Nn * Nn;
#pragma unroll
    for (int s = 0; s < 4; ++s)
        o[(size_t)(m0 + ty + 8 * s) * Nn + n0 + tx] = f2b(tile[tx][ty + 8 * s]);
}

__global__ void maxpool_k(const float* __restrict__ h, float* __restrict__ gf, int C, int Nn)
{
    const int c = blockIdx.x, b = blockIdx.y, t = threadIdx.x;
    const float* p = h + ((size_t)b * C + c) * Nn;
    float mx = -3.4e38f;
    for (int i = t; i < Nn; i += 256) mx = fmaxf(mx, p[i]);
    __shared__ float red[256];
    red[t] = mx; __syncthreads();
    for (int s = 128; s > 0; s >>= 1) { if (t < s) red[t] = fmaxf(red[t], red[t + s]); __syncthreads(); }
    if (t == 0) gf[b * C + c] = red[0];
}

// Point-major bf16 [B][N][128]: cols 0-63 = h, cols 64-127 = broadcast gf.
__global__ void concat_k(const float* __restrict__ h, const float* __restrict__ gf,
                         u16* __restrict__ out, int Nn)
{
    const int n = blockIdx.x * 256 + threadIdx.x, b = blockIdx.y;
    const float* hp = h + (size_t)b * 64 * Nn + n;
    u16* op = out + ((size_t)b * Nn + n) * 128;
    for (int c = 0; c < 64; ++c) op[c]      = f2b(hp[(size_t)c * Nn]);
    for (int c = 0; c < 64; ++c) op[64 + c] = f2b(gf[b * 64 + c]);
}

// global_feature: out[(b*N+n)*256 + c] = h[b][c][n].  LDS-tiled transpose.
__global__ __launch_bounds__(256)
void gfeat_k(const float* __restrict__ h, float* __restrict__ out, int Nn)
{
    __shared__ float tile[32][33];
    const int b = blockIdx.z;
    const int n0 = blockIdx.x * 32, c0 = blockIdx.y * 32;
    const int tx = threadIdx.x, ty = threadIdx.y;   // 32 x 8
    const float* hp = h + (size_t)b * 256 * Nn;
#pragma unroll
    for (int s = 0; s < 4; ++s)
        tile[ty + 8 * s][tx] = hp[(size_t)(c0 + ty + 8 * s) * Nn + n0 + tx];
    __syncthreads();
    float* op = out + (size_t)b * Nn * 256;
#pragma unroll
    for (int s = 0; s < 4; ++s)
        op[(size_t)(n0 + ty + 8 * s) * 256 + c0 + tx] = tile[tx][ty + 8 * s];
}

// ---------------------------------------------------------------------------
// Host side
// ---------------------------------------------------------------------------
static void run_sa(hipStream_t st, int C,
                   const float* hF, const u16* hBT,
                   const u16* Wqk, const u16* Wv, const float* bv,
                   const u16* Wt, const float* bnS, const float* bnSh,
                   float* outF,
                   float* attE, u16* attT, u16* qT, u16* xvb, u16* tbT,
                   float* colsum)
{
    const int B = 8, N = 2048;
    dim3 blk(128);
    // q = Wqk @ h, stored transposed bf16 qT[B][N][C]
    gemm_wmma<EPI_QT><<<dim3(N / 128, C / 64, B), blk, 0, st>>>(
        Wqk, 0, C, hBT, (long long)N * C, C,
        nullptr, qT, (long long)N * C, C,
        nullptr, nullptr, nullptr, 0, nullptr, C);
    // xv = Wv @ h + bv, bf16 row-major [B][C][N]
    gemm_wmma<EPI_BIAS_BF16><<<dim3(N / 128, C / 64, B), blk, 0, st>>>(
        Wv, 0, C, hBT, (long long)N * C, C,
        nullptr, xvb, (long long)C * N, N,
        bv, nullptr, nullptr, 0, nullptr, C);
    // energy[n][m] = sum_c qT[n][c] * qT[m][c]  (fp32 [B][N][N])
    gemm_wmma<EPI_ENERGY><<<dim3(N / 128, N / 64, B), blk, 0, st>>>(
        qT, (long long)N * C, C, qT, (long long)N * C, C,
        attE, nullptr, (long long)N * N, N,
        nullptr, nullptr, nullptr, 0, nullptr, C);
    softmax_rows<<<dim3(N, B), 256, 0, st>>>(attE, N);
    colsum_k<<<dim3(N / 256, B), 256, 0, st>>>(attE, colsum, N);
    attT_k<<<dim3(N / 32, N / 32, B), dim3(32, 8), 0, st>>>(attE, attT, N);
    // t = h - (xv@att)/(1e-9+colsum), stored transposed bf16 tbT[B][N][C]
    gemm_wmma<EPI_ATT_T><<<dim3(N / 128, C / 64, B), blk, 0, st>>>(
        xvb, (long long)C * N, N, attT, (long long)N * N, N,
        nullptr, tbT, (long long)C * N, N,
        nullptr, nullptr, hF, (long long)C * N, colsum, N);
    // out = h + relu(bn(Wt @ t + bt))
    gemm_wmma<EPI_BNRELU_RES><<<dim3(N / 128, C / 64, B), blk, 0, st>>>(
        Wt, 0, C, tbT, (long long)C * N, C,
        outF, nullptr, (long long)C * N, N,
        bnS, bnSh, hF, (long long)C * N, nullptr, C);
}

extern "C" void kernel_launch(void* const* d_in, const int* in_sizes, int n_in,
                              void* d_out, int out_size, void* d_ws, size_t ws_size,
                              hipStream_t stream)
{
    (void)in_sizes; (void)n_in; (void)out_size; (void)ws_size;
    const int B = 8, N = 2048;

    const float* x    = (const float*)d_in[0];
    const float* W1   = (const float*)d_in[1];
    const float* b1   = (const float*)d_in[2];
    const float* g1   = (const float*)d_in[3];
    const float* be1  = (const float*)d_in[4];
    const float* m1   = (const float*)d_in[5];
    const float* v1   = (const float*)d_in[6];
    const float* W2   = (const float*)d_in[7];
    const float* b2   = (const float*)d_in[8];
    const float* g2   = (const float*)d_in[9];
    const float* be2  = (const float*)d_in[10];
    const float* m2   = (const float*)d_in[11];
    const float* v2   = (const float*)d_in[12];
    const float* Wqk1 = (const float*)d_in[13];
    const float* Wv1  = (const float*)d_in[14];
    const float* bv1  = (const float*)d_in[15];
    const float* Wt1  = (const float*)d_in[16];
    const float* bt1  = (const float*)d_in[17];
    const float* ga1  = (const float*)d_in[18];
    const float* ba1  = (const float*)d_in[19];
    const float* ma1  = (const float*)d_in[20];
    const float* va1  = (const float*)d_in[21];
    const float* W3   = (const float*)d_in[22];
    const float* b3   = (const float*)d_in[23];
    const float* g3   = (const float*)d_in[24];
    const float* be3  = (const float*)d_in[25];
    const float* m3   = (const float*)d_in[26];
    const float* v3   = (const float*)d_in[27];
    const float* Wqk2 = (const float*)d_in[28];
    const float* Wv2  = (const float*)d_in[29];
    const float* bv2  = (const float*)d_in[30];
    const float* Wt2  = (const float*)d_in[31];
    const float* bt2  = (const float*)d_in[32];
    const float* ga2  = (const float*)d_in[33];
    const float* ba2  = (const float*)d_in[34];
    const float* ma2  = (const float*)d_in[35];
    const float* va2  = (const float*)d_in[36];

    // ---- workspace carve (256B aligned) ----
    char* w = (char*)d_ws;
    auto alloc = [&](size_t bytes) -> void* {
        void* p = (void*)w;
        w += (bytes + 255) & ~(size_t)255;
        return p;
    };
    float* attE   = (float*)alloc((size_t)B * N * N * 4);    // 134 MB
    u16*   attT   = (u16*)  alloc((size_t)B * N * N * 2);    // 67 MB
    float* fX     = (float*)alloc((size_t)B * 256 * N * 4);  // activations ping
    float* fY     = (float*)alloc((size_t)B * 256 * N * 4);  // activations pong
    u16*   bXT    = (u16*)  alloc((size_t)B * N * 256 * 2);  // bf16 SA input, point-major
    u16*   qT     = (u16*)  alloc((size_t)B * N * 256 * 2);
    u16*   xvb    = (u16*)  alloc((size_t)B * 256 * N * 2);  // row-major (A operand)
    u16*   tbT    = (u16*)  alloc((size_t)B * N * 256 * 2);  // point-major
    u16*   h128T  = (u16*)  alloc((size_t)B * N * 128 * 2);  // point-major
    float* colsum = (float*)alloc((size_t)B * N * 4);
    float* gf     = (float*)alloc((size_t)B * 64 * 4);
    u16*   Wqk1b  = (u16*)alloc(64 * 64 * 2);
    u16*   Wv1b   = (u16*)alloc(64 * 64 * 2);
    u16*   Wt1b   = (u16*)alloc(64 * 64 * 2);
    u16*   Wqk2b  = (u16*)alloc(256 * 256 * 2);
    u16*   Wv2b   = (u16*)alloc(256 * 256 * 2);
    u16*   Wt2b   = (u16*)alloc(256 * 256 * 2);
    u16*   W3b    = (u16*)alloc(256 * 128 * 2);
    float* sA1    = (float*)alloc(64 * 4);
    float* shA1   = (float*)alloc(64 * 4);
    float* s3     = (float*)alloc(256 * 4);
    float* sh3    = (float*)alloc(256 * 4);
    float* sA2    = (float*)alloc(256 * 4);
    float* shA2   = (float*)alloc(256 * 4);

    // ---- weight conversions + BN constant prep ----
    cvt_k<<<dim3(16), 256, 0, stream>>>(Wqk1, Wqk1b, 64 * 64);
    cvt_k<<<dim3(16), 256, 0, stream>>>(Wv1,  Wv1b,  64 * 64);
    cvt_k<<<dim3(16), 256, 0, stream>>>(Wt1,  Wt1b,  64 * 64);
    cvt_k<<<dim3(256), 256, 0, stream>>>(Wqk2, Wqk2b, 256 * 256);
    cvt_k<<<dim3(256), 256, 0, stream>>>(Wv2,  Wv2b,  256 * 256);
    cvt_k<<<dim3(256), 256, 0, stream>>>(Wt2,  Wt2b,  256 * 256);
    cvt_k<<<dim3(128), 256, 0, stream>>>(W3,   W3b,   256 * 128);
    bnprep_k<<<dim3(1), 256, 0, stream>>>(bt1, ga1, ba1, ma1, va1, sA1, shA1, 64);
    bnprep_k<<<dim3(1), 256, 0, stream>>>(b3,  g3,  be3, m3,  v3,  s3,  sh3,  256);
    bnprep_k<<<dim3(1), 256, 0, stream>>>(bt2, ga2, ba2, ma2, va2, sA2, shA2, 256);

    // ---- conv1+conv2 (+BN+ReLU) -> fX fp32 [B,64,N], bXT bf16 [B,N,64] ----
    conv12_k<<<dim3(N / 256, B), 256, 0, stream>>>(
        x, W1, b1, g1, be1, m1, v1, W2, b2, g2, be2, m2, v2, fX, bXT, N);

    // ---- SA1 (C=64): fX/bXT -> fY ----
    run_sa(stream, 64, fX, bXT, Wqk1b, Wv1b, bv1, Wt1b, sA1, shA1,
           fY, attE, attT, qT, xvb, tbT, colsum);

    // ---- maxpool + concat -> h128T bf16 [B,N,128] ----
    maxpool_k<<<dim3(64, B), 256, 0, stream>>>(fY, gf, 64, N);
    concat_k<<<dim3(N / 256, B), 256, 0, stream>>>(fY, gf, h128T, N);

    // ---- conv3 (128->256) + BN + ReLU -> fX fp32 + bXT bf16 [B,N,256] ----
    gemm_wmma<EPI_BNRELU><<<dim3(N / 128, 256 / 64, B), dim3(128), 0, stream>>>(
        W3b, 0, 128, h128T, (long long)N * 128, 128,
        fX, bXT, (long long)256 * N, N,
        s3, sh3, nullptr, 0, nullptr, 128);

    // ---- SA2 (C=256): fX/bXT -> d_out (h part) ----
    run_sa(stream, 256, fX, bXT, Wqk2b, Wv2b, bv2, Wt2b, sA2, shA2,
           (float*)d_out, attE, attT, qT, xvb, tbT, colsum);

    // ---- global_feature = h transposed -> second half of d_out ----
    gfeat_k<<<dim3(N / 32, 256 / 32, B), dim3(32, 8), 0, stream>>>(
        (const float*)d_out, (float*)d_out + (size_t)B * 256 * N, N);
}
